// CrossAttention_89189290868783
// MI455X (gfx1250) — compile-verified, matrix operand order checked by reading, not run on hardware
//
#include <hip/hip_runtime.h>
#include <cstdint>
#include <cstddef>

typedef __attribute__((ext_vector_type(16))) _Float16 v16h;
typedef __attribute__((ext_vector_type(8)))  _Float16 h8;
typedef __attribute__((ext_vector_type(8)))  float    v8f;

// Fixed problem dims from the reference
#define BATCH 2
#define SQ    2048
#define NH    32
#define DH    128
#define SK    2048
#define HKV   8

// ---------------- CDNA5 async global->LDS path (guarded) ----------------
#if defined(__HIP_DEVICE_COMPILE__) && __has_builtin(__builtin_amdgcn_global_load_async_to_lds_b128)
#define USE_ASYNC_LDS 1
#define AS1 __attribute__((address_space(1)))
#define AS3 __attribute__((address_space(3)))
typedef int v4i_vs __attribute__((vector_size(16)));   // matches builtin param pointee
__device__ __forceinline__ void async_copy16(const void* g, void* l) {
    __builtin_amdgcn_global_load_async_to_lds_b128((AS1 v4i_vs*)g, (AS3 v4i_vs*)l, 0, 0);
}
__device__ __forceinline__ void async_wait0() {
#if __has_builtin(__builtin_amdgcn_s_wait_asynccnt)
    __builtin_amdgcn_s_wait_asynccnt(0);
#else
    asm volatile("s_wait_asynccnt 0x0" ::: "memory");
#endif
}
#endif

// ---------------- shared flash-attention block compute ----------------
struct FaState {
    float m[8];
    float l[8];
    v8f   o[8];
};

template <bool MASKED>
__device__ __forceinline__ void fa_block_compute(const v16h qf[4],
                                                 _Float16 (*Ksh)[136],
                                                 _Float16 (*VshT)[40],
                                                 _Float16 (*Pw)[40],   // per-wave [16][40]
                                                 int kb, int qrow0, int lane,
                                                 FaState& st)
{
    const int half = lane >> 4;
    const int l16  = lane & 15;
    const v8f vzero = {};
    const float NEG_INF = -__builtin_inff();

    // ---- S = (Q*scale) K^T : two 16x16 tiles over the 32-wide K block ----
    v8f sacc[2];
    #pragma unroll
    for (int tc = 0; tc < 2; ++tc) {
        v8f s = vzero;
        #pragma unroll
        for (int c = 0; c < 4; ++c) {
            const int key  = tc * 16 + l16;
            const int doff = 32 * c + half * 8;
            const h8 lo = *(const h8*)&Ksh[key][doff];
            const h8 hi = *(const h8*)&Ksh[key][doff + 16];
            v16h bfr;
            #pragma unroll
            for (int i = 0; i < 8; ++i) { bfr[i] = lo[i]; bfr[i + 8] = hi[i]; }
            s = __builtin_amdgcn_wmma_f32_16x16x32_f16(false, qf[c], false, bfr,
                                                       (short)0, s, false, false);
        }
        sacc[tc] = s;
    }

    // ---- causal mask (diagonal blocks only) + online softmax; stage P ----
    #pragma unroll
    for (int r = 0; r < 8; ++r) {
        float x0 = sacc[0][r];
        float x1 = sacc[1][r];
        if (MASKED) {
            const int row = qrow0 + r + 8 * half;
            const int c0  = kb + l16;
            x0 = (c0 > row)      ? NEG_INF : x0;
            x1 = (c0 + 16 > row) ? NEG_INF : x1;
        }
        float mx = fmaxf(x0, x1);
        #pragma unroll
        for (int off = 1; off < 16; off <<= 1)
            mx = fmaxf(mx, __shfl_xor(mx, off, 32));
        const float mnew = fmaxf(st.m[r], mx);
        const float msc  = __expf(st.m[r] - mnew);   // 0 on first visit
        const float p0   = __expf(x0 - mnew);
        const float p1   = __expf(x1 - mnew);
        float sum = p0 + p1;
        #pragma unroll
        for (int off = 1; off < 16; off <<= 1)
            sum += __shfl_xor(sum, off, 32);
        st.l[r] = st.l[r] * msc + sum;
        st.m[r] = mnew;
        #pragma unroll
        for (int j = 0; j < 8; ++j) st.o[j][r] *= msc;
        Pw[r + 8 * half][l16]      = (_Float16)p0;
        Pw[r + 8 * half][l16 + 16] = (_Float16)p1;
    }

    // ---- O += P V ----
    v16h pf;
    {
        const int base = half * 8;
        const h8 lo = *(const h8*)&Pw[l16][base];
        const h8 hi = *(const h8*)&Pw[l16][base + 16];
        #pragma unroll
        for (int i = 0; i < 8; ++i) { pf[i] = lo[i]; pf[i + 8] = hi[i]; }
    }
    #pragma unroll
    for (int j = 0; j < 8; ++j) {
        const int drow = j * 16 + l16;
        const int koff = half * 8;
        const h8 lo = *(const h8*)&VshT[drow][koff];
        const h8 hi = *(const h8*)&VshT[drow][koff + 16];
        v16h bfr;
        #pragma unroll
        for (int i = 0; i < 8; ++i) { bfr[i] = lo[i]; bfr[i + 8] = hi[i]; }
        st.o[j] = __builtin_amdgcn_wmma_f32_16x16x32_f16(false, pf, false, bfr,
                                                         (short)0, st.o[j], false, false);
    }
}

__device__ __forceinline__ void fa_load_q(const float* __restrict__ q,
                                          int bz, int hy, int qrow0, int lane,
                                          v16h qf[4])
{
    const int half = lane >> 4;
    const int l16  = lane & 15;
    const float scale = 0.08838834764831845f; // 1/sqrt(128)
    const int trow = qrow0 + l16;
    const float* qp = q + (((size_t)bz * SQ + trow) * NH + hy) * DH;
    #pragma unroll
    for (int c = 0; c < 4; ++c) {
        const int base = 32 * c + half * 8;
        #pragma unroll
        for (int i = 0; i < 8; ++i) {
            qf[c][i]     = (_Float16)(qp[base + i]      * scale);
            qf[c][i + 8] = (_Float16)(qp[base + 16 + i] * scale);
        }
    }
}

__device__ __forceinline__ void fa_store_out(float* __restrict__ out,
                                             int bz, int hy, int qrow0, int lane,
                                             FaState& st)
{
    const int half = lane >> 4;
    const int l16  = lane & 15;
    float rl[8];
    #pragma unroll
    for (int r = 0; r < 8; ++r) rl[r] = 1.0f / st.l[r];
    #pragma unroll
    for (int j = 0; j < 8; ++j) {
        const int dcol = j * 16 + l16;
        #pragma unroll
        for (int r = 0; r < 8; ++r) {
            const int trow = qrow0 + r + 8 * half;
            out[(((size_t)bz * SQ + trow) * NH + hy) * DH + dcol] = st.o[j][r] * rl[r];
        }
    }
}

// ---------------- KV fp32 -> f16 preconversion (K row-major, V transposed) ----------------
__global__ __launch_bounds__(256)
void CrossAttention_89189290868783_kvprep(const float* __restrict__ kv,
                                          _Float16* __restrict__ K16,
                                          _Float16* __restrict__ V16T)
{
    __shared__ _Float16 Vt[DH][72];   // 144B row stride (16B aligned)
    const int tid = threadIdx.x;
    const int s0  = blockIdx.x * 64;
    const int bh  = blockIdx.y;            // b*HKV + kvh
    const int b   = bh / HKV;
    const int kvh = bh % HKV;

    {
        const int r  = tid >> 2;           // 0..63 local key row
        const int cg = (tid & 3) * 32;     // 32-wide column group
        const size_t base = (((size_t)b * SK + (s0 + r)) * 2) * (HKV * DH) +
                            (size_t)kvh * DH + cg;
        const float* kp = kv + base;                      // kv[...,0,kvh,:]
        const float* vp = kv + base + (size_t)HKV * DH;   // kv[...,1,kvh,:]
        _Float16* kout = K16 + ((size_t)bh * SK + (s0 + r)) * DH + cg;
        #pragma unroll
        for (int i = 0; i < 32; ++i) kout[i] = (_Float16)kp[i];
        #pragma unroll
        for (int i = 0; i < 32; ++i) Vt[cg + i][r] = (_Float16)vp[i];
    }
    __syncthreads();
    {
        const int d  = tid >> 1;           // 0..127
        const int sg = (tid & 1) * 32;
        _Float16* vout = V16T + ((size_t)bh * DH + d) * SK + s0 + sg;
        #pragma unroll
        for (int i = 0; i < 32; ++i) vout[i] = Vt[d][sg + i];
    }
}

// ---- fast path: double-buffered f16 tiles via async global->LDS, overlapped with WMMA ----
__global__ __launch_bounds__(128)
void CrossAttention_89189290868783_fast(const float* __restrict__ q,
                                        const _Float16* __restrict__ K16,
                                        const _Float16* __restrict__ V16T,
                                        float* __restrict__ out)
{
    __shared__ __align__(16) _Float16 Ksh[2][32][136];
    __shared__ __align__(16) _Float16 VshT[2][128][40];
    __shared__ __align__(16) _Float16 Psh[4][16][40];

    const int tid  = threadIdx.x;
    const int lane = tid & 31;
    const int wave = tid >> 5;

    const int bx  = blockIdx.x;
    const int hy  = blockIdx.y;
    const int bz  = blockIdx.z;
    const int bh  = bz * HKV + (hy >> 2);

    const int qrow0 = bx * 64 + wave * 16;
    const int kend  = (bx + 1) * 64;

    auto stage = [&](int kbs, int buf) {
        const int r32 = tid >> 2;
        const int cg  = (tid & 3) * 32;
        const _Float16* kg = K16 + ((size_t)bh * SK + kbs + r32) * DH + cg;
        const _Float16* vg = V16T + ((size_t)bh * DH + tid) * SK + kbs;
#ifdef USE_ASYNC_LDS
        #pragma unroll
        for (int c = 0; c < 4; ++c) async_copy16(kg + c * 8, &Ksh[buf][r32][cg + c * 8]);
        #pragma unroll
        for (int c = 0; c < 4; ++c) async_copy16(vg + c * 8, &VshT[buf][tid][c * 8]);
#else
        #pragma unroll
        for (int c = 0; c < 4; ++c)
            *(h8*)&Ksh[buf][r32][cg + c * 8] = *(const h8*)(kg + c * 8);
        #pragma unroll
        for (int c = 0; c < 4; ++c)
            *(h8*)&VshT[buf][tid][c * 8] = *(const h8*)(vg + c * 8);
#endif
        if (kbs + 32 < kend) {   // hint tile after next -> global_prefetch_b8
            __builtin_prefetch(kg + 32 * DH, 0, 1);
            __builtin_prefetch(vg + 32, 0, 1);
        }
    };

    v16h qf[4];
    fa_load_q(q, bz, hy, qrow0, lane, qf);

    FaState st;
    const v8f vzero = {};
    #pragma unroll
    for (int r = 0; r < 8; ++r) { st.m[r] = -__builtin_inff(); st.l[r] = 0.0f; }
    #pragma unroll
    for (int j = 0; j < 8; ++j) st.o[j] = vzero;

    stage(0, 0);                            // prologue fill of buffer 0

    int it = 0;
    for (int kb = 0; kb < kend; kb += 32, ++it) {
#ifdef USE_ASYNC_LDS
        async_wait0();                      // this wave's fills for buf[it&1] landed
#endif
        __syncthreads();                    // all waves filled; prev compute done
        if (kb + 32 < kend)
            stage(kb + 32, (it + 1) & 1);   // DMA next tile under this tile's math
        if (kb + 32 > qrow0)                // diagonal block for this wave
            fa_block_compute<true >(qf, Ksh[it & 1], VshT[it & 1], Psh[wave],
                                    kb, qrow0, lane, st);
        else                                // strictly-lower block: no mask needed
            fa_block_compute<false>(qf, Ksh[it & 1], VshT[it & 1], Psh[wave],
                                    kb, qrow0, lane, st);
    }

    fa_store_out(out, bz, hy, qrow0, lane, st);
}

// ---------------- fallback: fused fp32 KV conversion (single buffer) ----------------
__global__ __launch_bounds__(128)
void CrossAttention_89189290868783_kernel(const float* __restrict__ q,
                                          const float* __restrict__ kv,
                                          float* __restrict__ out)
{
    __shared__ __align__(16) _Float16 Ksh[32][136];
    __shared__ __align__(16) _Float16 VshT[128][40];
    __shared__ __align__(16) _Float16 Psh[4][16][40];

    const int tid  = threadIdx.x;
    const int lane = tid & 31;
    const int wave = tid >> 5;

    const int bx  = blockIdx.x;
    const int hy  = blockIdx.y;
    const int bz  = blockIdx.z;
    const int kvh = hy >> 2;

    const int qrow0 = bx * 64 + wave * 16;

    v16h qf[4];
    fa_load_q(q, bz, hy, qrow0, lane, qf);

    FaState st;
    const v8f vzero = {};
    #pragma unroll
    for (int r = 0; r < 8; ++r) { st.m[r] = -__builtin_inff(); st.l[r] = 0.0f; }
    #pragma unroll
    for (int j = 0; j < 8; ++j) st.o[j] = vzero;

    const int kend = (bx + 1) * 64;

    for (int kb = 0; kb < kend; kb += 32) {
        __syncthreads();
        {
            const int r32 = tid >> 2;
            const int cg  = (tid & 3) * 32;
            const size_t rowbase =
                ((size_t)bz * SK + (size_t)(kb + r32)) * (2 * HKV * DH) +
                (size_t)kvh * DH + cg;
            const float* kp = kv + rowbase;
            const float* vp = kv + rowbase + (size_t)HKV * DH;
            #pragma unroll
            for (int i = 0; i < 32; ++i) Ksh[r32][cg + i]  = (_Float16)kp[i];
            #pragma unroll
            for (int i = 0; i < 32; ++i) VshT[cg + i][r32] = (_Float16)vp[i];
            if (kb + 32 < kend) {
                __builtin_prefetch(kp + 2 * HKV * DH * 32, 0, 1);
                __builtin_prefetch(vp + 2 * HKV * DH * 32, 0, 1);
            }
        }
        __syncthreads();

        fa_block_compute<true>(qf, Ksh, VshT, Psh[wave], kb, qrow0, lane, st);
    }

    fa_store_out(out, bz, hy, qrow0, lane, st);
}

extern "C" void kernel_launch(void* const* d_in, const int* in_sizes, int n_in,
                              void* d_out, int out_size, void* d_ws, size_t ws_size,
                              hipStream_t stream) {
    (void)in_sizes; (void)n_in; (void)out_size;
    const float* q  = (const float*)d_in[0];
    const float* kv = (const float*)d_in[1];
    float* out = (float*)d_out;

    const size_t kbytes = (size_t)BATCH * HKV * SK * DH * 2;  // 8 MB f16 K
    dim3 grid(SQ / 64, NH, BATCH);
    dim3 block(128);

    if (ws_size >= 2 * kbytes) {
        _Float16* K16  = (_Float16*)d_ws;
        _Float16* V16T = (_Float16*)((char*)d_ws + kbytes);
        CrossAttention_89189290868783_kvprep<<<dim3(SK / 64, BATCH * HKV), 256, 0, stream>>>(
            kv, K16, V16T);
        CrossAttention_89189290868783_fast<<<grid, block, 0, stream>>>(q, K16, V16T, out);
    } else {
        CrossAttention_89189290868783_kernel<<<grid, block, 0, stream>>>(q, kv, out);
    }
}
